// NWHead_36103495090241
// MI455X (gfx1250) — matrix operand order repaired and checked by reading, not compiled, not run on hardware
//
#include <hip/hip_runtime.h>
#include <math.h>

// ---------------------------------------------------------------------------
// NWHead on MI455X (gfx1250, wave32, WMMA + Tensor Data Mover).
//
// Roofline: support_x (134MB) + support_y (262MB) reads dominate (~17us at
// 23.3 TB/s HBM). Projection GEMM (8.6 GFLOP) runs on v_wmma_f32_16x16x32_f16
// with f32 accumulation; A tiles are DMA'd by the TDM (tensor_load_to_lds)
// into double-buffered LDS, overlapped with compute; B (= W, 128 KB) lives in
// registers per wave for the whole kernel. support_y is streamed with
// non-temporal loads since it exceeds the 192 MB L2.
// ---------------------------------------------------------------------------

typedef __attribute__((ext_vector_type(16))) _Float16 v16h;
typedef __attribute__((ext_vector_type(2)))  __fp16   vp2;   // cvt_pkrtz result type
typedef __attribute__((ext_vector_type(8)))  float    v8f;
typedef __attribute__((ext_vector_type(2)))  float    vf2;
typedef unsigned v4u __attribute__((ext_vector_type(4)));
typedef int      v4i __attribute__((ext_vector_type(4)));
typedef int      v8i __attribute__((ext_vector_type(8)));

#define B_ 8
#define S_ 8192
#define F_ 512
#define E_ 128
#define C_ 1000

// Padded A-tile layout: element e (= m*512+k) lives at dword e + 2*(e>>8)
// (TDM pad: 2 dwords every 256 dwords -> effective row stride 516 dwords).
#define ATILE_PAD 8256   // dwords per buffer (8192 data + pads, rounded)

#if defined(__has_builtin)
#if __has_builtin(__builtin_amdgcn_tensor_load_to_lds)
#define HAVE_TDM 1
#endif
#endif
#if __has_include(<hip/amd_detail/amd_gfx1250_TDM.h>)
#define TDM_6ARG 1
#endif

#if defined(HAVE_TDM)
// One TDM descriptor load: 2D tile, 16 rows x 512 f32, row stride 512 elems,
// LDS padding 2 dwords per 256 dwords. Issued by one wave; EXEC is ignored.
__device__ __forceinline__ void tile_issue_wave(const float* gsrc, unsigned lds_off) {
  unsigned long long ga = (unsigned long long)gsrc;
  v4u g0 = { 1u,                                   // count=1 (valid, user mode)
             lds_off,                              // lds_addr (bytes)
             (unsigned)ga,                         // global_addr[31:0]
             (unsigned)((ga >> 32) & 0x01FFFFFFull) | (2u << 30) }; // [56:32] | type=2
  union { v8i v; unsigned u[8]; } g1;
  g1.u[0] = (2u << 16)        // data_size = 4 bytes
          | (1u << 20)        // pad_enable
          | (7u << 22)        // pad_interval: every 256 dwords
          | (1u << 25);       // pad_amount: 2 dwords
  g1.u[1] = (512u & 0xFFFFu) << 16;            // tensor_dim0 lo16 = 512
  g1.u[2] = (512u >> 16) | (16u << 16);        // dim0 hi | tensor_dim1 lo16 = 16
  g1.u[3] = (16u >> 16) | (512u << 16);        // dim1 hi | tile_dim0 = 512
  g1.u[4] = 16u;                               // tile_dim1 = 16, tile_dim2 = 0
  g1.u[5] = 512u;                              // tensor_dim0_stride lo32 = 512
  g1.u[6] = 0u;
  g1.u[7] = 0u;
  v4i z4 = {0, 0, 0, 0};
#if defined(TDM_6ARG)
  v8i z8 = {0, 0, 0, 0, 0, 0, 0, 0};
  __builtin_amdgcn_tensor_load_to_lds(g0, g1.v, z4, z4, z8, 0);
#else
  __builtin_amdgcn_tensor_load_to_lds(g0, g1.v, z4, z4, 0);
#endif
}
#define TILE_ISSUE(g, l, tid) do { if ((tid) < 32) tile_issue_wave((g), (l)); } while (0)
#define TILE_WAIT(tid)        do { if ((tid) < 32) __builtin_amdgcn_s_wait_tensorcnt(0); } while (0)
#else
// Fallback: per-lane async global->LDS copies (same padded layout), b64 chunks.
__device__ __forceinline__ void tile_issue_all(const float* gsrc, unsigned lds_off, int tid) {
  for (int r = 0; r < 16; ++r) {
    int e0 = (r * 256 + tid) * 2;                       // even element index
    unsigned loff = lds_off + (unsigned)(e0 + 2 * (e0 >> 8)) * 4u;
    const float* gp = gsrc + e0;
    asm volatile("global_load_async_to_lds_b64 %0, %1, off"
                 :: "v"(loff), "v"(gp) : "memory");
  }
}
#define TILE_ISSUE(g, l, tid) tile_issue_all((g), (l), (tid))
#define TILE_WAIT(tid)        asm volatile("s_wait_asynccnt 0x0" ::: "memory")
#endif

// ---------------- prep: W (F x E f32) -> WhT (E x F f16) -------------------
__global__ __launch_bounds__(256) void prep_wht(const float* __restrict__ W,
                                                _Float16* __restrict__ WhT) {
  int idx = blockIdx.x * 256 + threadIdx.x;     // 65536 threads
  int e = idx >> 9;
  int f = idx & 511;
  WhT[idx] = (_Float16)W[f * E_ + e];
}

// ---------------- xq = x @ W  (8 x 128, f32, tiny) -------------------------
__global__ __launch_bounds__(128) void xq_kernel(const float* __restrict__ x,
                                                 const float* __restrict__ W,
                                                 float* __restrict__ xq) {
  int b = blockIdx.x;
  int e = threadIdx.x;
  float acc = 0.f;
  for (int f = 0; f < F_; ++f)
    acc += x[b * F_ + f] * W[f * E_ + e];
  xq[b * E_ + e] = acc;
}

// ---------------- main: sx = support_x @ W, fused -||xq - sx|| -------------
// grid (64, 8): blockIdx.x = s-chunk of 128 rows, blockIdx.y = b.
// 256 threads = 8 waves; wave w owns embedding columns n0 = 16*w .. +15.
// A tiles: TDM double-buffered LDS. B: registers (128 dwords/lane), no LDS.
__global__ __launch_bounds__(256) void gemm_dist(
    const float* __restrict__ supx, const _Float16* __restrict__ WhT,
    const float* __restrict__ xq, float* __restrict__ scores) {
  __shared__ __align__(16) unsigned Abuf[2][ATILE_PAD];

  const int tid  = threadIdx.x;
  const int lane = tid & 31;
  const int wave = tid >> 5;
  const int b    = blockIdx.y;
  const int n0   = wave * 16;
  const int nl   = lane & 15;
  const int hi   = lane >> 4;

  const unsigned lds0 = (unsigned)(unsigned long long)(const void*)&Abuf[0][0];
  const unsigned lds1 = (unsigned)(unsigned long long)(const void*)&Abuf[1][0];
  const float* gtile0 = supx + ((size_t)b * S_ + (size_t)blockIdx.x * 128) * F_;

  // Kick off DMA of tile 0 immediately, then load B while it flies.
  TILE_ISSUE(gtile0, lds0, tid);

  // B fragments for all 16 k-steps, resident in VGPRs for the whole kernel.
  // B 32x16 f16 layout: n = n0+nl, k = 8*(lane>>4) + 16*(j>=4) + 2*(j&3).
  unsigned breg[16][8];
  const unsigned* gw = (const unsigned*)WhT;
#pragma unroll
  for (int ks = 0; ks < 16; ++ks) {
    int base = ((n0 + nl) * F_ + ks * 32 + hi * 8) >> 1;   // dword index
#pragma unroll
    for (int j = 0; j < 8; ++j) breg[ks][j] = gw[base + (j & 3) + ((j >= 4) ? 8 : 0)];
  }
  const float xqv = xq[b * E_ + n0 + nl];

  // Per-lane fragment base (vf2 units). A 16x32 f16 layout gives element
  // e = nl*516 + hi*8 + Kc with Kc = ks*32 + 16*(j>=4) + 2*(j&3) + 2*(ks>=8)
  // (pad term 2*(k>>8) is constant per k-step: k never crosses a 256 line).
  // -> single address VGPR + compile-time DS immediate offsets.
  const int fragbase = nl * 258 + hi * 4;

  for (int mt = 0; mt < 8; ++mt) {
    TILE_WAIT(tid);
    __syncthreads();                         // buf[mt&1] ready for everyone
    if (mt < 7)                              // prefetch next tile (overlapped)
      TILE_ISSUE(gtile0 + (size_t)(mt + 1) * 16 * F_, (mt & 1) ? lds0 : lds1, tid);

    const vf2* As2 = (const vf2*)Abuf[mt & 1] + fragbase;
    v8f acc = {};
#pragma unroll
    for (int ks = 0; ks < 16; ++ks) {
      union { v16h h; vp2 p[8]; } a;
      union { v16h h; unsigned u[8]; } bm;
#pragma unroll
      for (int j = 0; j < 8; ++j) {
        // constant vf2 offset per (ks, j): ds_load_b64 with immediate offset
        int off = ks * 16 + ((j >= 4) ? 8 : 0) + (j & 3) + ((ks >= 8) ? 1 : 0);
        vf2 p = As2[off];                             // ds_load_b64 (f32 pair)
        a.p[j] = __builtin_amdgcn_cvt_pkrtz(p.x, p.y);
        bm.u[j] = breg[ks][j];
      }
      acc = __builtin_amdgcn_wmma_f32_16x16x32_f16(
          false, a.h, false, bm.h, (short)0, acc, false, false);
    }

    // Epilogue: squared distance. Reduce over 16 lanes (same n-group) with
    // shuffles, combine the 8 wave n-tiles through LDS (reuse dead buf head).
    __syncthreads();                         // all waves done reading buf[mt&1]
    float* sdist = (float*)&Abuf[mt & 1][0];
    if (tid < 16) sdist[tid] = 0.f;
    __syncthreads();
#pragma unroll
    for (int j = 0; j < 8; ++j) {
      float d = acc[j] - xqv;                // C/D: row m = hi*8 + j, col n = nl
      float t = d * d;
      t += __shfl_xor(t, 1, 32);
      t += __shfl_xor(t, 2, 32);
      t += __shfl_xor(t, 4, 32);
      t += __shfl_xor(t, 8, 32);
      if (nl == 0) atomicAdd(&sdist[hi * 8 + j], t);
    }
    __syncthreads();
    if (tid < 16)
      scores[(size_t)b * S_ + (size_t)(blockIdx.x * 8 + mt) * 16 + tid] =
          -sqrtf(sdist[tid]);
  }
}

// ---------------- softmax over S, in place ---------------------------------
__global__ __launch_bounds__(256) void softmax_k(float* __restrict__ sc_all) {
  __shared__ float red[256];
  const int b = blockIdx.x;
  const int tid = threadIdx.x;
  float* sc = sc_all + (size_t)b * S_;

  float mx = -1e30f;
  for (int i = tid; i < S_; i += 256) mx = fmaxf(mx, sc[i]);
  red[tid] = mx; __syncthreads();
  for (int o = 128; o > 0; o >>= 1) {
    if (tid < o) red[tid] = fmaxf(red[tid], red[tid + o]);
    __syncthreads();
  }
  mx = red[0]; __syncthreads();

  float sum = 0.f;
  for (int i = tid; i < S_; i += 256) {
    float e = __expf(sc[i] - mx);
    sc[i] = e;
    sum += e;
  }
  red[tid] = sum; __syncthreads();
  for (int o = 128; o > 0; o >>= 1) {
    if (tid < o) red[tid] += red[tid + o];
    __syncthreads();
  }
  float inv = 1.f / red[0];
  for (int i = tid; i < S_; i += 256) sc[i] *= inv;
}

// ---------------- out = probs @ support_y, deterministic two-stage ---------
__global__ __launch_bounds__(256) void wsum_partial(
    const float* __restrict__ probs, const float* __restrict__ sy,
    float* __restrict__ partials) {
  const int tid = threadIdx.x;
  const int schunk = blockIdx.x, ct = blockIdx.y, b = blockIdx.z;
  const int c = ct * 256 + tid;
  float acc = 0.f;
  if (c < C_) {
    const float* p = probs + (size_t)b * S_;
    const float* y = sy + (size_t)b * S_ * C_;
    const int sbeg = schunk * 512;
    for (int s = sbeg; s < sbeg + 512; ++s)
      acc += p[s] * __builtin_nontemporal_load(&y[(size_t)s * C_ + c]);
  }
  partials[((size_t)b * 16 + schunk) * 1024 + c] = acc;
}

__global__ __launch_bounds__(256) void wsum_final(
    const float* __restrict__ partials, float* __restrict__ out) {
  const int tid = threadIdx.x;
  const int ct = blockIdx.x, b = blockIdx.y;
  const int c = ct * 256 + tid;
  if (c >= C_) return;
  float acc = 0.f;
  for (int k = 0; k < 16; ++k)
    acc += partials[((size_t)b * 16 + k) * 1024 + c];
  out[(size_t)b * C_ + c] = acc;
}

// ---------------------------------------------------------------------------
extern "C" void kernel_launch(void* const* d_in, const int* in_sizes, int n_in,
                              void* d_out, int out_size, void* d_ws, size_t ws_size,
                              hipStream_t stream) {
  const float* x    = (const float*)d_in[0];  // (8, 512)
  const float* supx = (const float*)d_in[1];  // (8, 8192, 512)
  const float* supy = (const float*)d_in[2];  // (8, 8192, 1000)
  const float* W    = (const float*)d_in[3];  // (1, 512, 128)
  float* out        = (float*)d_out;          // (8, 1000)

  char* ws = (char*)d_ws;
  _Float16* WhT   = (_Float16*)(ws);          // 128*512*2   = 128 KB
  float*    xq    = (float*)(ws + 131072);    // 8*128*4     =   4 KB
  float*    sc    = (float*)(ws + 135168);    // 8*8192*4    = 256 KB
  float*    parts = (float*)(ws + 397312);    // 8*16*1024*4 = 512 KB

  prep_wht    <<<dim3(256),       256, 0, stream>>>(W, WhT);
  xq_kernel   <<<dim3(8),         128, 0, stream>>>(x, W, xq);
  gemm_dist   <<<dim3(64, 8),     256, 0, stream>>>(supx, WhT, xq, sc);
  softmax_k   <<<dim3(8),         256, 0, stream>>>(sc);
  wsum_partial<<<dim3(16, 4, 8),  256, 0, stream>>>(sc, supy, parts);
  wsum_final  <<<dim3(4, 8),      256, 0, stream>>>(parts, out);
}
